// OptimizedMCLPModel_84069689852052
// MI455X (gfx1250) — matrix operand order repaired and checked by reading, not compiled, not run on hardware
//
#include <hip/hip_runtime.h>

// ---------------------------------------------------------------------------
// MI455X (gfx1250) GNN forward pass.
//
// Roofline: dense FLOPs (~5 GF) are trivial vs the bf16 WMMA ceiling; the
// edge scatter (~350MB + 34M f32 atomics) bounds the model at ~20us of HBM
// traffic (23.3 TB/s).  All GEMMs use v_wmma_f32_16x16x32_bf16 (bf16 in /
// f32 accumulate keeps the f32 exponent range of N(0,1)-scale activations).
// Weight panels are stored transposed in bf16 and staged into LDS with the
// Tensor Data Mover (tensor_load_to_lds + s_wait_tensorcnt) so every WMMA
// fragment load is a b128 (global_load_b128 / ds_load_b128) — no scalar
// 16-bit gathers in any hot loop.  Attention is fused flash-style (268MB
// score tensor never touches HBM) on bf16 Q/K/Vt written directly by the
// GEMM epilogue (1/sqrt(DH) folded into Q, V stored transposed).
// ---------------------------------------------------------------------------

#define N_NODES 4096
#define N_EDGES 262144
#define OUT_F   64
#define N_HEADS 4

typedef __bf16 bf16;
typedef __attribute__((ext_vector_type(16))) __bf16        v16bf;
typedef __attribute__((ext_vector_type(8)))  float         v8f;
typedef __attribute__((ext_vector_type(4)))  unsigned int  v4u;
typedef __attribute__((ext_vector_type(8)))  int           v8i;
typedef __attribute__((ext_vector_type(4)))  int           v4i;

union U16 { v16bf v; uint4 q[2]; };

#define GF_BIAS  1
#define GF_RELU  2
#define GF_RES   4
#define GF_BF16  8
#define GF_TRANS 16

// ---------------------------------------------------------------------------
// WMMA GEMM: C[M x Nc] = act(A[M x K] @ W[K x Nc] + bias) (+res) (*oscale)
// W is pre-converted bf16 *transposed* (Wt[Nc][K]) in global; the whole
// panel (<=128x192 = 48KB) is DMA'd into LDS once per block by the Tensor
// Data Mover (all 8 waves share it).  One wave = one 16x16 tile, K step 32.
// A/B fragments use contiguous k (kb+16h+s); with Wt, a B-fragment for lane `col` is
// Wt[col][kb+16h .. +15]: 16 contiguous bf16 -> two ds_load_b128.
// ---------------------------------------------------------------------------
__global__ void k_gemm_wmma(const float* __restrict__ A, const bf16* __restrict__ Wt,
                            const float* __restrict__ bias, const float* __restrict__ res,
                            float* __restrict__ C,
                            int M, int K, int Nc, int ldc, int col_off, int flags,
                            float oscale)
{
    __shared__ bf16 Wsh[192 * 128];            // 48KB worst case (Wm1)

    // ---- TDM: stage bf16 Wt panel [Nc x K] into LDS (wave 0 issues DMA) ---
    if (threadIdx.x < 32) {
        unsigned lds = (unsigned)(size_t)(&Wsh[0]);   // flat LDS low32 = LDS offset
        unsigned long long ga = (unsigned long long)(size_t)Wt;
        unsigned d0 = (unsigned)K, d1 = (unsigned)Nc;            // dim0=K, dim1=Nc
        unsigned long long s0 = (unsigned long long)K;           // row stride = K
        unsigned long long s1 = (unsigned long long)K * (unsigned long long)Nc;
        // D# group0: count=1 | lds_addr | global_addr[56:0] | type=2 (ISA 8.3)
        v4u g0 = { 1u, lds,
                   (unsigned)(ga & 0xffffffffu),
                   (unsigned)((ga >> 32) & 0x01ffffffu) | 0x80000000u };
        // D# group1: data_size=2B; tensor dims; tile dims; dim0 stride (ISA 8.4)
        v8i g1 = { (int)(1u << 16),
                   (int)((d0 & 0xffffu) << 16),
                   (int)(((d0 >> 16) & 0xffffu) | ((d1 & 0xffffu) << 16)),
                   (int)(((d1 >> 16) & 0xffffu) | ((d0 & 0xffffu) << 16)),   // tile_dim0 = K
                   (int)(d1 & 0xffffu),                                      // tile_dim1 = Nc
                   (int)(unsigned)(s0 & 0xffffffffu),
                   (int)((unsigned)((s0 >> 32) & 0xffffu) | ((unsigned)(s1 & 0xffffu) << 16)),
                   (int)(unsigned)((s1 >> 16) & 0xffffffffu) };
        v4i g2 = { 0, 0, 0, 0 };
        v4i g3 = { 0, 0, 0, 0 };
#if defined(__clang_major__) && (__clang_major__ >= 23)
        v8i g4 = { 0, 0, 0, 0, 0, 0, 0, 0 };
        __builtin_amdgcn_tensor_load_to_lds(g0, g1, g2, g3, g4, 0);
#else
        __builtin_amdgcn_tensor_load_to_lds(g0, g1, g2, g3, 0);
#endif
        __builtin_amdgcn_s_wait_tensorcnt((short)0);
    }
    __syncthreads();

    int wave = (int)((blockIdx.x * blockDim.x + threadIdx.x) >> 5);
    int lane = threadIdx.x & 31;
    int tilesN = Nc >> 4;
    int tm = wave / tilesN;
    int tn = wave - tm * tilesN;
    int r = lane & 15;
    int h = lane >> 4;
    if (tm * 16 >= M) return;                  // wave-uniform (grids are exact)

    v8f acc = {0.f, 0.f, 0.f, 0.f, 0.f, 0.f, 0.f, 0.f};
    const float* Arow = A + (size_t)(tm * 16 + r) * K;
    const bf16*  Wrow = Wsh + (size_t)(tn * 16 + r) * K;   // this lane's Wt row
    for (int kb = 0; kb < K; kb += 32) {
        __builtin_prefetch(Arow + kb + 64, 0, 0);          // global_prefetch_b8
        v16bf af;
        // A-frag: two contiguous 8-float runs -> b128 loads, cvt to bf16
        const float4* a0 = (const float4*)(Arow + kb + 16 * h);
        const float4* a1 = (const float4*)(Arow + kb + 16 * h + 8);
        float4 x0 = a0[0], x1 = a0[1], x2 = a1[0], x3 = a1[1];
        af[0]  = (bf16)x0.x; af[1]  = (bf16)x0.y; af[2]  = (bf16)x0.z; af[3]  = (bf16)x0.w;
        af[4]  = (bf16)x1.x; af[5]  = (bf16)x1.y; af[6]  = (bf16)x1.z; af[7]  = (bf16)x1.w;
        af[8]  = (bf16)x2.x; af[9]  = (bf16)x2.y; af[10] = (bf16)x2.z; af[11] = (bf16)x2.w;
        af[12] = (bf16)x3.x; af[13] = (bf16)x3.y; af[14] = (bf16)x3.z; af[15] = (bf16)x3.w;
        // B-frag: 16 contiguous bf16 from the TDM-staged LDS panel -> 2x ds_load_b128
        U16 bu;
        const bf16* Wp = Wrow + kb + 16 * h;
        bu.q[0] = *(const uint4*)(Wp);
        bu.q[1] = *(const uint4*)(Wp + 8);
        acc = __builtin_amdgcn_wmma_f32_16x16x32_bf16(false, af, false, bu.v,
                                                      (short)0, acc, false, false);
    }
    int col = tn * 16 + r;
    float b = (flags & GF_BIAS) ? bias[col] : 0.f;
#pragma unroll
    for (int i = 0; i < 8; ++i) {
        int row = tm * 16 + i + 8 * h;
        float v = acc[i] + b;
        if (flags & GF_RELU) v = fmaxf(v, 0.f);
        if (flags & GF_RES)  v += res[(size_t)row * Nc + col];
        v *= oscale;
        if (flags & GF_TRANS)     ((bf16*)C)[(size_t)(col_off + col) * ldc + row] = (bf16)v;
        else if (flags & GF_BF16) ((bf16*)C)[(size_t)row * ldc + col_off + col] = (bf16)v;
        else                      C[(size_t)row * ldc + col_off + col] = v;
    }
}

// ---------------------------------------------------------------------------
// Flash attention on bf16 Q/K (row-major) and Vt (transposed [64 x n]).
// Wave = (16-query block, one head); 32 keys/iter: two score WMMAs (DH=16
// padded to 32) + one P@V WMMA (K=32, no padding).  All fragment loads are
// aligned b128.  Online softmax row stats in registers via shfl_xor over the
// 16 lanes holding each row in the C layout; P transposed C->A layout
// through a 1KB per-wave LDS tile (per-wave LDS is in-order: no barrier).
// ---------------------------------------------------------------------------
__global__ void k_attn(const bf16* __restrict__ Q, const bf16* __restrict__ Km,
                       const bf16* __restrict__ Vt, float* __restrict__ O, int n)
{
    __shared__ bf16 Pld[8][16 * 32];
    int wv   = threadIdx.x >> 5;
    int lane = threadIdx.x & 31;
    int gw   = blockIdx.x * 8 + wv;
    int head = gw & (N_HEADS - 1);
    int qb   = gw >> 2;
    int r = lane & 15, h = lane >> 4;
    int hoff = head * 16;

    // Q fragment: lanes h==0 hold DH 0..15 (two aligned b128); h==1 = K-dim padding
    U16 qu;
    const bf16* qrow = Q + (size_t)(qb * 16 + r) * OUT_F + hoff;
    qu.q[0] = h ? make_uint4(0u, 0u, 0u, 0u) : *(const uint4*)(qrow);
    qu.q[1] = h ? make_uint4(0u, 0u, 0u, 0u) : *(const uint4*)(qrow + 8);
    v16bf qf = qu.v;

    const bf16* vrow = Vt + (size_t)(hoff + r) * n;   // this lane's Vt row

    v8f o = {0.f, 0.f, 0.f, 0.f, 0.f, 0.f, 0.f, 0.f};
    float mrun[8], lrun[8];
#pragma unroll
    for (int i = 0; i < 8; ++i) { mrun[i] = -3.0e38f; lrun[i] = 0.f; }
    bf16* P = &Pld[wv][0];

    for (int kb = 0; kb < n; kb += 32) {
        U16 k0u, k1u;
        if (h == 0) {                          // real DH rows; h==1 half is pad
            const bf16* k0 = Km + (size_t)(kb + r) * OUT_F + hoff;
            const bf16* k1 = Km + (size_t)(kb + 16 + r) * OUT_F + hoff;
            k0u.q[0] = *(const uint4*)(k0);
            k0u.q[1] = *(const uint4*)(k0 + 8);
            k1u.q[0] = *(const uint4*)(k1);
            k1u.q[1] = *(const uint4*)(k1 + 8);
        } else {
            k0u.q[0] = k0u.q[1] = make_uint4(0u, 0u, 0u, 0u);
            k1u.q[0] = k1u.q[1] = make_uint4(0u, 0u, 0u, 0u);
        }
        v8f z = {0.f, 0.f, 0.f, 0.f, 0.f, 0.f, 0.f, 0.f};
        v8f s0 = __builtin_amdgcn_wmma_f32_16x16x32_bf16(false, qf, false, k0u.v, (short)0, z, false, false);
        v8f s1 = __builtin_amdgcn_wmma_f32_16x16x32_bf16(false, qf, false, k1u.v, (short)0, z, false, false);

        float t[8], mnew[8], al[8], ps[8];
#pragma unroll
        for (int i = 0; i < 8; ++i) t[i] = fmaxf(s0[i], s1[i]);
#pragma unroll
        for (int m = 1; m < 16; m <<= 1)
#pragma unroll
            for (int i = 0; i < 8; ++i) t[i] = fmaxf(t[i], __shfl_xor(t[i], m, 32));
#pragma unroll
        for (int i = 0; i < 8; ++i) {
            mnew[i] = fmaxf(mrun[i], t[i]);
            al[i]   = __expf(mrun[i] - mnew[i]);
            float p0 = __expf(s0[i] - mnew[i]);
            float p1 = __expf(s1[i] - mnew[i]);
            ps[i] = p0 + p1;
            s0[i] = p0; s1[i] = p1;
            o[i] *= al[i];
            mrun[i] = mnew[i];
        }
#pragma unroll
        for (int m = 1; m < 16; m <<= 1)
#pragma unroll
            for (int i = 0; i < 8; ++i) ps[i] += __shfl_xor(ps[i], m, 32);
#pragma unroll
        for (int i = 0; i < 8; ++i) lrun[i] = lrun[i] * al[i] + ps[i];

        // C-layout -> A-layout transpose of P through LDS
#pragma unroll
        for (int i = 0; i < 8; ++i) {
            P[(i + 8 * h) * 32 + r]      = (bf16)s0[i];
            P[(i + 8 * h) * 32 + 16 + r] = (bf16)s1[i];
        }
        U16 pu, vu;
        pu.q[0] = *(const uint4*)(P + r * 32 + 16 * h);       // ds_load_b128
        pu.q[1] = *(const uint4*)(P + r * 32 + 16 * h + 8);
        const bf16* vp = vrow + kb + 16 * h;                  // global b128 x2
        vu.q[0] = *(const uint4*)(vp);
        vu.q[1] = *(const uint4*)(vp + 8);
        o = __builtin_amdgcn_wmma_f32_16x16x32_bf16(false, pu.v, false, vu.v, (short)0, o, false, false);
    }
#pragma unroll
    for (int i = 0; i < 8; ++i)
        O[(size_t)(qb * 16 + i + 8 * h) * OUT_F + hoff + r] = o[i] / lrun[i];
}

// --------------------------- small helper kernels ---------------------------
__global__ void k_zero(float* p, int n)
{
    int i = blockIdx.x * blockDim.x + threadIdx.x;
    if (i < n) p[i] = 0.f;
}

// f32 W[K x Nc] -> bf16 Wt[Nc x K] (transpose + convert)
__global__ void k_cvt_t(const float* __restrict__ in, bf16* __restrict__ out,
                        int K, int Nc)
{
    int idx = blockIdx.x * blockDim.x + threadIdx.x;
    if (idx < K * Nc) {
        int k = idx / Nc, nn = idx - k * Nc;
        out[(size_t)nn * K + k] = (bf16)in[idx];
    }
}

__global__ void k_deg(const int* __restrict__ dst, const float* __restrict__ ew,
                      float* deg, int E)
{
    int i = blockIdx.x * blockDim.x + threadIdx.x;
    if (i < E) atomicAdd(&deg[dst[i]], ew[i]);
}

__global__ void k_dinv(const float* deg, float* dinv, int n)
{
    int i = blockIdx.x * blockDim.x + threadIdx.x;
    if (i < n) {
        float d = deg[i] + 1.0f;               // + self-loop weight
        dinv[i] = (d > 0.f) ? rsqrtf(d) : 0.f;
    }
}

// scatter-add:  AGG[dst] += dinv[src]*ew*dinv[dst] * H[src]
__global__ void k_edge_agg(const int* __restrict__ src, const int* __restrict__ dst,
                           const float* __restrict__ ew, const float* __restrict__ dinv,
                           const float* __restrict__ H, float* AGG, int E, int fshift)
{
    long long tot    = (long long)E << fshift;
    long long stride = (long long)gridDim.x * blockDim.x;
    int mask = (1 << fshift) - 1;
    for (long long idx = blockIdx.x * (long long)blockDim.x + threadIdx.x;
         idx < tot; idx += stride) {
        int e = (int)(idx >> fshift);
        int f = (int)idx & mask;
        int s = src[e], d = dst[e];
        float nm = dinv[s] * ew[e] * dinv[d];
        atomicAdd(&AGG[((size_t)d << fshift) + f], nm * H[((size_t)s << fshift) + f]);
    }
}

// self-loop (norm = dinv^2, ew = 1) plus GCN bias
__global__ void k_self_bias(const float* __restrict__ H, const float* __restrict__ dinv,
                            const float* __restrict__ b, float* AGG, int M, int F)
{
    int idx = blockIdx.x * blockDim.x + threadIdx.x;
    if (idx < M * F) {
        int i = idx / F, f = idx - i * F;
        float dv = dinv[i];
        AGG[idx] += dv * dv * H[idx] + b[f];
    }
}

__global__ void k_bn_stats(const float* __restrict__ X, float* mean, float* var,
                           int M, int F)
{
    __shared__ float s1[256], s2[256];
    int f = blockIdx.x, t = threadIdx.x;
    float a = 0.f, b = 0.f;
    for (int i = t; i < M; i += 256) {
        float v = X[(size_t)i * F + f];
        a += v; b += v * v;
    }
    s1[t] = a; s2[t] = b;
    __syncthreads();
    for (int o = 128; o > 0; o >>= 1) {
        if (t < o) { s1[t] += s1[t + o]; s2[t] += s2[t + o]; }
        __syncthreads();
    }
    if (t == 0) {
        float m = s1[0] / (float)M;
        mean[f] = m;
        var[f]  = s2[0] / (float)M - m * m;
    }
}

__global__ void k_bn_relu(float* X, const float* __restrict__ mean, const float* __restrict__ var,
                          const float* __restrict__ g, const float* __restrict__ be, int M, int F)
{
    int idx = blockIdx.x * blockDim.x + threadIdx.x;
    if (idx < M * F) {
        int f = idx % F;
        float v = X[idx];
        float r = g[f] * (v - mean[f]) * rsqrtf(var[f] + 1e-5f) + be[f];
        X[idx] = fmaxf(r, 0.f);
    }
}

// relu(s*W1+b1) @ W2 + b2 for scalar input s; writes a column slice of the
// "combined" buffer (concat assembled in place).
__global__ void k_mlp2_scalar(const float* __restrict__ inp, const float* __restrict__ W1,
                              const float* __restrict__ b1, const float* __restrict__ W2,
                              const float* __restrict__ b2, float* out, int ldc, int col_off)
{
    __shared__ float hid[64];
    int nidx = blockIdx.x, t = threadIdx.x;
    float s = inp[nidx];
    hid[t] = fmaxf(s * W1[t] + b1[t], 0.f);
    __syncthreads();
    float acc = b2[t];
#pragma unroll
    for (int j = 0; j < 64; ++j) acc += hid[j] * W2[j * 64 + t];
    out[(size_t)nidx * ldc + col_off + t] = acc;
}

// final 64->1 head; mode 1 = sigmoid
__global__ void k_head1(const float* __restrict__ X, const float* __restrict__ w,
                        const float* __restrict__ b, float* out, int n, int Kd, int mode)
{
    int i = blockIdx.x * blockDim.x + threadIdx.x;
    if (i >= n) return;
    float acc = b[0];
    for (int k = 0; k < Kd; ++k) acc += X[(size_t)i * Kd + k] * w[k];
    if (mode == 1) acc = 1.f / (1.f + __expf(-acc));
    out[i] = acc;
}

// ---------------------------------------------------------------------------
extern "C" void kernel_launch(void* const* d_in, const int* in_sizes, int n_in,
                              void* d_out, int out_size, void* d_ws, size_t ws_size,
                              hipStream_t stream)
{
    (void)in_sizes; (void)n_in; (void)out_size; (void)ws_size;
    const float* x    = (const float*)d_in[0];
    const int*   ei   = (const int*)d_in[1];
    const int*   src  = ei;
    const int*   dst  = ei + N_EDGES;
    const float* ew   = (const float*)d_in[2];
    const float* dist = (const float*)d_in[3];
    const float* degf = (const float*)d_in[4];
    const float* const* pp = (const float* const*)(d_in + 5);
    enum { W1, b1, W2, b2, W3, b3, g1, be1, g2, be2, g3, be3,
           Wd1, bd1, Wd2, bd2, Wg1, bg1, Wg2, bg2,
           Wq, bq, Wk, bk, Wv, bv, Wo, bo,
           Wm1, bm1, Wm2, bm2, Wr, br,
           Wf1, bf1, Wf2, bf2, Wf3, bf3, Wc1, bc1, Wc2, bc2 };

    // workspace layout (float offsets); total ~15MB
    float* ws   = (float*)d_ws;
    float* DEG  = ws;
    float* DINV = ws + 4096;
    float* MEAN = ws + 8192;
    float* VARB = ws + 8448;
    float* BH   = ws + 16384;          // 4096x128 lin output
    float* BA   = BH + 524288;         // 4096x128 agg / current h
    float* BQ   = BA + 524288;         // bf16 Q region
    float* BK   = BQ + 262144;         // bf16 K region
    float* BV   = BK + 262144;         // bf16 Vt region
    float* BATT = BV + 262144;         // 4096x64 attention out (pre-Wo)
    float* BCMB = BATT + 262144;       // 4096x192 combined
    float* BT1  = BCMB + 786432;       // 4096x128 temp
    float* XRES = BT1 + 524288;        // 4096x64
    bf16*  wbf  = (bf16*)(XRES + 262144);  // bf16 transposed weights (~232KB)
    float* emb  = (float*)d_out;
    float* fac  = emb + 262144;
    float* cov  = fac + 4096;

    bf16* QB16 = (bf16*)BQ;
    bf16* KB16 = (bf16*)BK;
    bf16* VT16 = (bf16*)BV;

    // bf16 transposed-weight carve-out
    bf16 *W1b = wbf;          wbf += 16384;
    bf16 *W2b = wbf;          wbf += 16384;
    bf16 *W3b = wbf;          wbf += 8192;
    bf16 *Wrb = wbf;          wbf += 8192;
    bf16 *Wqb = wbf;          wbf += 4096;
    bf16 *Wkb = wbf;          wbf += 4096;
    bf16 *Wvb = wbf;          wbf += 4096;
    bf16 *Wob = wbf;          wbf += 4096;
    bf16 *Wm1b = wbf;         wbf += 24576;
    bf16 *Wm2b = wbf;         wbf += 8192;
    bf16 *Wf1b = wbf;         wbf += 8192;
    bf16 *Wf2b = wbf;         wbf += 8192;
    bf16 *Wc1b = wbf;

    auto cvt = [&](int pI, bf16* dstp, int K, int Nc) {
        int n = K * Nc;
        k_cvt_t<<<(n + 255) / 256, 256, 0, stream>>>(pp[pI], dstp, K, Nc);
    };
    cvt(W1, W1b, 128, 128);  cvt(W2, W2b, 128, 128);  cvt(W3, W3b, 128, 64);
    cvt(Wr, Wrb, 128, 64);   cvt(Wq, Wqb, 64, 64);    cvt(Wk, Wkb, 64, 64);
    cvt(Wv, Wvb, 64, 64);    cvt(Wo, Wob, 64, 64);    cvt(Wm1, Wm1b, 192, 128);
    cvt(Wm2, Wm2b, 128, 64); cvt(Wf1, Wf1b, 64, 128); cvt(Wf2, Wf2b, 128, 64);
    cvt(Wc1, Wc1b, 64, 64);

    auto gemm = [&](const float* A, const bf16* Wp, int pB, int Nc, int K, float* C,
                    int ldc, int coff, int flags, const float* res, float osc) {
        int tiles = (N_NODES / 16) * (Nc / 16);      // always a multiple of 8
        k_gemm_wmma<<<tiles / 8, 256, 0, stream>>>(A, Wp, pB >= 0 ? pp[pB] : nullptr,
                                                   res, C, N_NODES, K, Nc, ldc, coff,
                                                   flags, osc);
    };

    // ---- degrees / symmetric normalization --------------------------------
    k_zero<<<16, 256, 0, stream>>>(DEG, 4096);
    k_deg<<<(N_EDGES + 255) / 256, 256, 0, stream>>>(dst, ew, DEG, N_EDGES);
    k_dinv<<<16, 256, 0, stream>>>(DEG, DINV, 4096);

    // ---- residual path ----------------------------------------------------
    gemm(x, Wrb, br, 64, 128, XRES, 64, 0, GF_BIAS, nullptr, 1.f);

    // ---- 3 GCN layers: lin -> scatter agg -> self-loop+bias -> BN -> relu -
    auto gcn = [&](const float* hin, const bf16* Wp, int pB, int pG, int pBe, int F) {
        int fshift = (F == 128) ? 7 : 6;
        gemm(hin, Wp, -1, F, 128, BH, F, 0, 0, nullptr, 1.f);
        k_zero<<<(N_NODES * F + 255) / 256, 256, 0, stream>>>(BA, N_NODES * F);
        k_edge_agg<<<2048, 256, 0, stream>>>(src, dst, ew, DINV, BH, BA, N_EDGES, fshift);
        k_self_bias<<<(N_NODES * F + 255) / 256, 256, 0, stream>>>(BH, DINV, pp[pB], BA, N_NODES, F);
        k_bn_stats<<<F, 256, 0, stream>>>(BA, MEAN, VARB, N_NODES, F);
        k_bn_relu<<<(N_NODES * F + 255) / 256, 256, 0, stream>>>(BA, MEAN, VARB, pp[pG], pp[pBe], N_NODES, F);
    };
    gcn(x,  W1b, b1, g1, be1, 128);
    gcn(BA, W2b, b2, g2, be2, 128);
    gcn(BA, W3b, b3, g3, be3, 64);     // h3 = BA [4096x64]

    // ---- full-graph attention (flash-style, bf16 Q/K row-major, V transposed)
    gemm(BA, Wqb, bq, 64, 64, (float*)QB16, 64,      0, GF_BIAS | GF_BF16,  nullptr, 0.25f);
    gemm(BA, Wkb, bk, 64, 64, (float*)KB16, 64,      0, GF_BIAS | GF_BF16,  nullptr, 1.f);
    gemm(BA, Wvb, bv, 64, 64, (float*)VT16, N_NODES, 0, GF_BIAS | GF_TRANS, nullptr, 1.f);
    k_attn<<<128, 256, 0, stream>>>(QB16, KB16, VT16, BATT, N_NODES);

    // ---- combined = [h_attn@Wo+bo | d_feat | dg_feat] assembled in place --
    gemm(BATT, Wob, bo, 64, 64, BCMB, 192, 0, GF_BIAS, nullptr, 1.f);
    k_mlp2_scalar<<<N_NODES, 64, 0, stream>>>(dist, pp[Wd1], pp[bd1], pp[Wd2], pp[bd2], BCMB, 192, 64);
    k_mlp2_scalar<<<N_NODES, 64, 0, stream>>>(degf, pp[Wg1], pp[bg1], pp[Wg2], pp[bg2], BCMB, 192, 128);

    // ---- emb = mlp2(combined) + x_res -------------------------------------
    gemm(BCMB, Wm1b, bm1, 128, 192, BT1, 128, 0, GF_BIAS | GF_RELU, nullptr, 1.f);
    gemm(BT1,  Wm2b, bm2, 64,  128, emb, 64,  0, GF_BIAS | GF_RES, XRES, 1.f);

    // ---- fac head ---------------------------------------------------------
    gemm(emb, Wf1b, bf1, 128, 64,  BH,  128, 0, GF_BIAS | GF_RELU, nullptr, 1.f);
    gemm(BH,  Wf2b, bf2, 64,  128, BT1, 64,  0, GF_BIAS | GF_RELU, nullptr, 1.f);
    k_head1<<<16, 256, 0, stream>>>(BT1, pp[Wf3], pp[bf3], fac, N_NODES, 64, 0);

    // ---- cov head ---------------------------------------------------------
    gemm(emb, Wc1b, bc1, 64, 64, BH, 64, 0, GF_BIAS | GF_RELU, nullptr, 1.f);
    k_head1<<<16, 256, 0, stream>>>(BH, pp[Wc2], pp[bc2], cov, N_NODES, 64, 1);
}
